// CrossAttention_32796370273147
// MI455X (gfx1250) — matrix operand order repaired and checked by reading, compiled, not run on hardware
//
#include <hip/hip_runtime.h>
#include <hip/hip_bf16.h>

// ---------------------------------------------------------------------------
// Problem constants (from reference): N=8192 rows, D=512 features, B=64 batches
// ---------------------------------------------------------------------------
#define NROW 8192
#define DDIM 512

typedef __attribute__((ext_vector_type(16))) __bf16 bf16x16;
typedef __attribute__((ext_vector_type(8)))  __bf16 bf16x8;
typedef __attribute__((ext_vector_type(8)))  float  v8f;

union BfPack { bf16x16 v; bf16x8 h[2]; };

// ---------------------------------------------------------------------------
// Kernel: zero-fill (float4) for the attn output region
// ---------------------------------------------------------------------------
__global__ void zero_kernel(float4* __restrict__ p, size_t n4) {
    size_t i = (size_t)blockIdx.x * blockDim.x + threadIdx.x;
    if (i < n4) p[i] = make_float4(0.f, 0.f, 0.f, 0.f);
}

// ---------------------------------------------------------------------------
// Kernel: float -> bf16 conversion
// ---------------------------------------------------------------------------
__global__ void cvt_kernel(const float* __restrict__ in, __bf16* __restrict__ out, int n) {
    int i = blockIdx.x * blockDim.x + threadIdx.x;
    if (i < n) out[i] = (__bf16)in[i];
}

// ---------------------------------------------------------------------------
// Kernel: per-row block bounds via binary search on sorted batch[]
// ---------------------------------------------------------------------------
__global__ void bounds_kernel(const int* __restrict__ batch, int* __restrict__ rs,
                              int* __restrict__ re, int n) {
    int i = blockIdx.x * blockDim.x + threadIdx.x;
    if (i >= n) return;
    int b = batch[i];
    int lo = 0, hi = n;
    while (lo < hi) { int mid = (lo + hi) >> 1; if (batch[mid] < b) lo = mid + 1; else hi = mid; }
    rs[i] = lo;
    lo = 0; hi = n;
    while (lo < hi) { int mid = (lo + hi) >> 1; if (batch[mid] <= b) lo = mid + 1; else hi = mid; }
    re[i] = lo;
}

// ---------------------------------------------------------------------------
// Kernel: LayerNorm over D=512, one row per 256-thread block, OUT = bf16|float
// ---------------------------------------------------------------------------
template <typename OUT>
__global__ void ln_kernel(const float* __restrict__ x, const float* __restrict__ w,
                          const float* __restrict__ bb, OUT* __restrict__ out) {
    int row = blockIdx.x, tid = threadIdx.x;
    const float* xr = x + (size_t)row * DDIM;
    float v0 = xr[tid], v1 = xr[tid + 256];
    __shared__ float red[256];
    red[tid] = v0 + v1;
    __syncthreads();
    for (int s = 128; s > 0; s >>= 1) { if (tid < s) red[tid] += red[tid + s]; __syncthreads(); }
    float mu = red[0] * (1.0f / DDIM);
    __syncthreads();
    float d0 = v0 - mu, d1 = v1 - mu;
    red[tid] = d0 * d0 + d1 * d1;
    __syncthreads();
    for (int s = 128; s > 0; s >>= 1) { if (tid < s) red[tid] += red[tid + s]; __syncthreads(); }
    float rstd = rsqrtf(red[0] * (1.0f / DDIM) + 1e-5f);
    out[(size_t)row * DDIM + tid]       = (OUT)(d0 * rstd * w[tid] + bb[tid]);
    out[(size_t)row * DDIM + tid + 256] = (OUT)(d1 * rstd * w[tid + 256] + bb[tid + 256]);
}

// ---------------------------------------------------------------------------
// Generic WMMA GEMM: out[M,Nn] = A[M,K](bf16) @ W[Nn,K]^T (bf16)
//   optional bias add, optional SiLU, outputs to f32 and/or bf16 buffers.
// Block = 256 threads = 8 waves; wave -> one 16x16 tile; block tile 16 x 128.
// A lane layout (16x32 bf16): lane L row M=L%16, K chunks {k0..k0+7, k0+16..k0+23}, k0=(L<16?0:8)
// B lane layout (32x16 bf16): lane L col N=L%16, K chunks {k0..k0+7, k0+16..k0+23}, k0=(L<16?0:8)
// ---------------------------------------------------------------------------
__global__ __launch_bounds__(256)
void gemm_abt_kernel(const __bf16* __restrict__ A, const __bf16* __restrict__ W,
                     const float* __restrict__ bias, float* __restrict__ outF,
                     __bf16* __restrict__ outB, int K, int Nn, int act) {
    int tid  = threadIdx.x;
    int lane = tid & 31;
    int wv   = tid >> 5;
    int t0   = blockIdx.x * 16;
    int nb   = blockIdx.y * 128 + wv * 16;

    int arow = t0 + (lane & 15);
    int brow = nb + (lane & 15);
    int k0   = (lane < 16) ? 0 : 8;
    int kb0  = (lane < 16) ? 0 : 16;

    const __bf16* ap = A + (size_t)arow * K;
    const __bf16* wp = W + (size_t)brow * K;

    v8f acc = {};
#pragma unroll 4
    for (int kt = 0; kt < K; kt += 32) {
        BfPack a, b;
        a.h[0] = *(const bf16x8*)(ap + kt + k0);
        a.h[1] = *(const bf16x8*)(ap + kt + k0 + 16);
        b.h[0] = *(const bf16x8*)(wp + kt + k0);
        b.h[1] = *(const bf16x8*)(wp + kt + k0 + 16);
        acc = __builtin_amdgcn_wmma_f32_16x16x32_bf16(false, a.v, false, b.v,
                                                      (short)0, acc, false, false);
    }

    int ncol = nb + (lane & 15);
    float bv = bias ? bias[ncol] : 0.0f;
#pragma unroll
    for (int r = 0; r < 8; ++r) {
        int m = r + ((lane < 16) ? 0 : 8);
        float v = acc[r] + bv;
        if (act) v = v / (1.0f + __expf(-v));           // SiLU
        size_t o = (size_t)(t0 + m) * Nn + ncol;
        if (outF) outF[o] = v;
        if (outB) outB[o] = (__bf16)v;
    }
}

// ---------------------------------------------------------------------------
// Attention kernel: block-diagonal q@k^T, masked softmax, writes attn (fp32).
// One block per 16 rows; 8 waves stride over 16-wide column tiles in the
// union block range [S,E). Exact reference semantics: max includes 0.
// ---------------------------------------------------------------------------
__global__ __launch_bounds__(256)
void attn_kernel(const __bf16* __restrict__ qb, const __bf16* __restrict__ kb,
                 const int* __restrict__ rs, const int* __restrict__ re,
                 float* __restrict__ attn) {
    __shared__ int      srs[16], sre[16];
    __shared__ unsigned smax[16];
    __shared__ float    ssum[16];
    __shared__ int      sSE[2];

    int t0 = blockIdx.x * 16;
    int tid = threadIdx.x, lane = tid & 31, wv = tid >> 5;

    if (tid < 16) { srs[tid] = rs[t0 + tid]; sre[tid] = re[t0 + tid];
                    smax[tid] = 0u; ssum[tid] = 0.0f; }
    if (tid == 0) { sSE[0] = rs[t0] & ~15; sSE[1] = (re[t0 + 15] + 15) & ~15; }
    __syncthreads();
    int S = sSE[0], E = sSE[1];

    int arow = t0 + (lane & 15);
    int k0   = (lane < 16) ? 0 : 8;
    int kb0  = (lane < 16) ? 0 : 16;
    const __bf16* ap = qb + (size_t)arow * DDIM;

    // ---- phase 1: WMMA score tiles, masked write, row max (max includes 0) ----
    for (int jt = S + wv * 16; jt < E; jt += 128) {
        const __bf16* bp = kb + (size_t)(jt + (lane & 15)) * DDIM;
        v8f acc = {};
#pragma unroll 4
        for (int kt = 0; kt < DDIM; kt += 32) {
            BfPack a, b;
            a.h[0] = *(const bf16x8*)(ap + kt + k0);
            a.h[1] = *(const bf16x8*)(ap + kt + k0 + 16);
            b.h[0] = *(const bf16x8*)(bp + kt + k0);
            b.h[1] = *(const bf16x8*)(bp + kt + k0 + 16);
            acc = __builtin_amdgcn_wmma_f32_16x16x32_bf16(false, a.v, false, b.v,
                                                          (short)0, acc, false, false);
        }
        int ncol = jt + (lane & 15);
#pragma unroll
        for (int r = 0; r < 8; ++r) {
            int m = r + ((lane < 16) ? 0 : 8);
            bool in = (ncol >= srs[m]) && (ncol < sre[m]);
            float v = in ? acc[r] : 0.0f;
            attn[(size_t)(t0 + m) * NROW + ncol] = v;
            if (in && v > 0.0f) atomicMax(&smax[m], __float_as_uint(v));
        }
    }
    __threadfence();
    __syncthreads();

    // ---- phase 2: exp(dot - max), accumulate per-row sums ----
    int range = E - S;
    for (int idx = tid; idx < range * 16; idx += 256) {
        int m = idx / range;
        int j = S + (idx - m * range);
        if (j >= srs[m] && j < sre[m]) {
            size_t o = (size_t)(t0 + m) * NROW + j;
            float e = __expf(attn[o] - __uint_as_float(smax[m]));
            attn[o] = e;
            atomicAdd(&ssum[m], e);
        }
    }
    __threadfence();
    __syncthreads();

    // ---- phase 3: normalize ----
    for (int idx = tid; idx < range * 16; idx += 256) {
        int m = idx / range;
        int j = S + (idx - m * range);
        if (j >= srs[m] && j < sre[m]) {
            size_t o = (size_t)(t0 + m) * NROW + j;
            attn[o] = attn[o] / ssum[m];
        }
    }
}

// ---------------------------------------------------------------------------
// x_weighted = attn @ x_conf, restricted to the block-diagonal K range.
// A = attn (fp32 -> bf16 in regs), B = x_conf rows (needs column access ->
// staged through LDS). Block: 16 rows x 128 cols of D; 8 waves.
// ---------------------------------------------------------------------------
__global__ __launch_bounds__(256)
void xw_kernel(const float* __restrict__ attn, const __bf16* __restrict__ xcb,
               const int* __restrict__ rs, const int* __restrict__ re,
               __bf16* __restrict__ outB) {
    __shared__ __bf16 Bs[32 * 128];
    __shared__ int    sSE[2];

    int t0 = blockIdx.x * 16;
    int cb = blockIdx.y * 128;
    int tid = threadIdx.x, lane = tid & 31, wv = tid >> 5;

    if (tid == 0) { sSE[0] = rs[t0] & ~31; sSE[1] = (re[t0 + 15] + 31) & ~31; }
    __syncthreads();
    int S = sSE[0], E = sSE[1];

    int arow = t0 + (lane & 15);
    int k0   = (lane < 16) ? 0 : 8;
    int kb0  = (lane < 16) ? 0 : 16;
    int nloc = wv * 16 + (lane & 15);

    int sj = tid >> 3;          // staging row 0..31
    int sc = (tid & 7) * 16;    // staging col base

    v8f acc = {};
    for (int jt = S; jt < E; jt += 32) {
        __syncthreads();
        const __bf16* src = xcb + (size_t)(jt + sj) * DDIM + cb + sc;
        *(bf16x8*)&Bs[sj * 128 + sc]     = *(const bf16x8*)(src);
        *(bf16x8*)&Bs[sj * 128 + sc + 8] = *(const bf16x8*)(src + 8);
        __syncthreads();

        BfPack a, b;
        const float* ap = attn + (size_t)arow * NROW + jt + k0;
#pragma unroll
        for (int t = 0; t < 8; ++t) {
            a.v[t]     = (__bf16)ap[t];
            a.v[t + 8] = (__bf16)ap[t + 16];
        }
#pragma unroll
        for (int t = 0; t < 16; ++t) b.v[t] = Bs[(k0 + t + (t >= 8 ? 8 : 0)) * 128 + nloc];

        acc = __builtin_amdgcn_wmma_f32_16x16x32_bf16(false, a.v, false, b.v,
                                                      (short)0, acc, false, false);
    }

    int ncol = cb + nloc;
#pragma unroll
    for (int r = 0; r < 8; ++r) {
        int m = r + ((lane < 16) ? 0 : 8);
        outB[(size_t)(t0 + m) * DDIM + ncol] = (__bf16)acc[r];
    }
}

// ---------------------------------------------------------------------------
// Host-side launch
// ---------------------------------------------------------------------------
extern "C" void kernel_launch(void* const* d_in, const int* in_sizes, int n_in,
                              void* d_out, int out_size, void* d_ws, size_t ws_size,
                              hipStream_t stream) {
    const int N = NROW, D = DDIM;
    const float* x_mole = (const float*)d_in[0];
    const float* x_conf = (const float*)d_in[1];
    const float* W1     = (const float*)d_in[2];
    const float* W2     = (const float*)d_in[3];
    const float* phi1_w = (const float*)d_in[4];
    const float* phi1_b = (const float*)d_in[5];
    const float* phi2_w = (const float*)d_in[6];
    const float* phi2_b = (const float*)d_in[7];
    const float* rho_w1 = (const float*)d_in[8];
    const float* rho_b1 = (const float*)d_in[9];
    const float* rho_w2 = (const float*)d_in[10];
    const float* rho_b2 = (const float*)d_in[11];
    const float* rln_w  = (const float*)d_in[12];
    const float* rln_b  = (const float*)d_in[13];
    const int*   batch  = (const int*)d_in[14];

    float* x_enc = (float*)d_out;                     // [N, D]
    float* attn  = (float*)d_out + (size_t)N * D;     // [N, N]

    // workspace layout (~51 MB)
    char* ws = (char*)d_ws;
    __bf16* qb  = (__bf16*)ws;                        // N*D bf16
    __bf16* kb  = qb  + (size_t)N * D;
    __bf16* xcb = kb  + (size_t)N * D;                // x_conf as bf16
    __bf16* tmp = xcb + (size_t)N * D;                // LN output / x_weighted
    __bf16* wb  = tmp + (size_t)N * D;                // D*D weight staging
    float*  h2  = (float*)(wb + (size_t)D * D);       // N*D f32
    int*    rs  = (int*)(h2 + (size_t)N * D);
    int*    re  = rs + N;

    // 1. zero attn region (harness poisons d_out)
    {
        size_t n4 = (size_t)N * N / 4;
        zero_kernel<<<dim3((unsigned)((n4 + 255) / 256)), 256, 0, stream>>>((float4*)attn, n4);
    }
    // 2. per-row block bounds
    bounds_kernel<<<N / 256, 256, 0, stream>>>(batch, rs, re, N);

    // 3. q = LN(x_mole; phi1) @ W1^T
    cvt_kernel<<<(D * D + 255) / 256, 256, 0, stream>>>(W1, wb, D * D);
    ln_kernel<__bf16><<<N, 256, 0, stream>>>(x_mole, phi1_w, phi1_b, tmp);
    gemm_abt_kernel<<<dim3(N / 16, D / 128), 256, 0, stream>>>(tmp, wb, nullptr, nullptr, qb, D, D, 0);

    // 4. k = LN(x_conf; phi2) @ W2^T
    cvt_kernel<<<(D * D + 255) / 256, 256, 0, stream>>>(W2, wb, D * D);
    ln_kernel<__bf16><<<N, 256, 0, stream>>>(x_conf, phi2_w, phi2_b, tmp);
    gemm_abt_kernel<<<dim3(N / 16, D / 128), 256, 0, stream>>>(tmp, wb, nullptr, nullptr, kb, D, D, 0);

    // 5. x_conf -> bf16 for attn @ x_conf
    cvt_kernel<<<(N * D + 255) / 256, 256, 0, stream>>>(x_conf, xcb, N * D);

    // 6. block-diagonal masked softmax attention -> attn (fp32 in d_out)
    attn_kernel<<<N / 16, 256, 0, stream>>>(qb, kb, rs, re, attn);

    // 7. x_weighted (bf16) = attn @ x_conf   (into tmp)
    xw_kernel<<<dim3(N / 16, D / 128), 256, 0, stream>>>(attn, xcb, rs, re, tmp);

    // 8. h1 = SiLU(x_weighted @ rho_w1^T + b1)  -> reuse qb as h1 (bf16)
    cvt_kernel<<<(D * D + 255) / 256, 256, 0, stream>>>(rho_w1, wb, D * D);
    gemm_abt_kernel<<<dim3(N / 16, D / 128), 256, 0, stream>>>(tmp, wb, rho_b1, nullptr, qb, D, D, 1);

    // 9. h2 = SiLU(h1 @ rho_w2^T + b2)  (fp32 for final LN)
    cvt_kernel<<<(D * D + 255) / 256, 256, 0, stream>>>(rho_w2, wb, D * D);
    gemm_abt_kernel<<<dim3(N / 16, D / 128), 256, 0, stream>>>(qb, wb, rho_b2, h2, nullptr, D, D, 1);

    // 10. x_encoded = LN(h2; rho_ln) -> d_out
    ln_kernel<float><<<N, 256, 0, stream>>>(h2, rln_w, rln_b, x_enc);

    (void)n_in; (void)in_sizes; (void)out_size; (void)ws_size;
}
